// MILModelWithPositionalEncoding_65747359367324
// MI455X (gfx1250) — matrix operand order repaired and checked by reading, not compiled, not run on hardware
//
#include <hip/hip_runtime.h>
#include <math.h>

// Problem constants (match reference setup_inputs)
#define BB 4
#define NN 4096
#define DD 1536
#define HH 512
#define KK 8
#define R2C (400.0f * 400.0f)

typedef __attribute__((ext_vector_type(16))) __bf16 v16bf;
typedef __attribute__((ext_vector_type(8)))  __bf16 v8bf;
typedef __attribute__((ext_vector_type(8)))  float  v8f;

union V16 { v16bf v; v8bf h[2]; };

__device__ __forceinline__ v16bf load_frag16(const __bf16* p0, const __bf16* p1) {
    V16 u;
    u.h[0] = *(const v8bf*)p0;
    u.h[1] = *(const v8bf*)p1;
    return u.v;
}

// A-tile LDS lane stride: 16 bf16 payload + 8 bf16 pad = 48B (kills 4-way bank conflicts)
#define ALANE 24

// ---------------------------------------------------------------------------
// K0: anchor inverse norms + zero accumulators
// ---------------------------------------------------------------------------
__global__ void init_kernel(const float* __restrict__ anchors,
                            float* __restrict__ anchor_invn,
                            float* __restrict__ latent_acc,
                            float* __restrict__ concept_acc) {
    int tid = threadIdx.x;
    if (tid < KK) {
        float ss = 0.f;
        for (int d = 0; d < DD; ++d) { float v = anchors[tid * DD + d]; ss += v * v; }
        anchor_invn[tid] = 1.0f / fmaxf(sqrtf(ss), 1e-12f);
    }
    for (int i = tid; i < BB * HH; i += blockDim.x) latent_acc[i] = 0.f;
    if (tid < BB * KK) concept_acc[tid] = 0.f;
}

// ---------------------------------------------------------------------------
// K1: fused cosine-anchor scores + images fp32 -> bf16 (one pass over images)
// ---------------------------------------------------------------------------
__global__ void concept_cvt_kernel(const float* __restrict__ images,
                                   const float* __restrict__ anchors,
                                   const float* __restrict__ anchor_invn,
                                   __bf16* __restrict__ img_bf,
                                   float* __restrict__ concept_acc) {
    __shared__ float red[9 * 256];
    int row = blockIdx.x;              // 0..B*N-1
    int b   = row / NN;
    int tid = threadIdx.x;
    const float* x = images + (size_t)row * DD;
    float ss = 0.f, dot[KK];
#pragma unroll
    for (int k = 0; k < KK; ++k) dot[k] = 0.f;
    for (int d = tid; d < DD; d += 256) {
        float v = x[d];
        img_bf[(size_t)row * DD + d] = (__bf16)v;
        ss += v * v;
#pragma unroll
        for (int k = 0; k < KK; ++k) dot[k] += v * anchors[k * DD + d];
    }
    red[tid] = ss;
#pragma unroll
    for (int k = 0; k < KK; ++k) red[(k + 1) * 256 + tid] = dot[k];
    __syncthreads();
    for (int s = 128; s > 0; s >>= 1) {
        if (tid < s) {
#pragma unroll
            for (int q = 0; q < 9; ++q) red[q * 256 + tid] += red[q * 256 + tid + s];
        }
        __syncthreads();
    }
    if (tid < KK) {
        float invn = 1.0f / fmaxf(sqrtf(red[0]), 1e-12f);
        float sc   = red[(tid + 1) * 256] * invn * anchor_invn[tid];
        atomicAdd(&concept_acc[b * KK + tid], sc);
    }
}

// ---------------------------------------------------------------------------
// K2: weights -> transposed bf16 (rows contiguous along K for B-fragments)
// ---------------------------------------------------------------------------
__global__ void cvt_weights_kernel(const float* __restrict__ Wf,
                                   const float* __restrict__ Wg,
                                   const float* __restrict__ Ws,
                                   __bf16* __restrict__ WfT,
                                   __bf16* __restrict__ WgT,
                                   __bf16* __restrict__ WsT) {
    int idx = blockIdx.x * blockDim.x + threadIdx.x;
    if (idx < HH * DD) {
        int h = idx / DD, d = idx % DD;
        WfT[idx] = (__bf16)Wf[d * HH + h];
    }
    if (idx < HH * HH) {
        int ho = idx / HH, hi = idx % HH;
        WgT[idx] = (__bf16)Wg[hi * HH + ho];
        WsT[idx] = (__bf16)Ws[hi * HH + ho];
    }
}

// ---------------------------------------------------------------------------
// K3: feat GEMM (M=B*N, K=D, N=H), 64 rows x 256 cols per block.
// Each wave: 4 row-tiles x 2 col-tiles (8 accumulators) so every B fragment
// is reused 4x -> 4x less WfT L2 traffic. Fused bias+positional epilogue;
// writes feat bf16 (row-major) and relu(feat)^T bf16.
// ---------------------------------------------------------------------------
__global__ void gemm_feat_kernel(const __bf16* __restrict__ imgbf,
                                 const __bf16* __restrict__ WfT,
                                 const float* __restrict__ b_feat,
                                 const float* __restrict__ W_pos,
                                 const float* __restrict__ positions,
                                 __bf16* __restrict__ featbf,
                                 __bf16* __restrict__ hT) {
    int cb = blockIdx.x & 1;
    int rb = blockIdx.x >> 1;
    int b  = rb >> 6;                 // rb / (NN/64)
    int n0 = (rb & 63) << 6;          // 64-row stripe
    int tid = threadIdx.x;
    int wave = tid >> 5, lane = tid & 31;
    int lmod = lane & 15, lhalf = lane >> 4;
    int kbA = lhalf * 8;
    int kbB = lhalf * 8;
    int h_base = (cb << 8) + wave * 32;

    v8f acc[8];
#pragma unroll
    for (int a = 0; a < 8; ++a)
#pragma unroll
        for (int j = 0; j < 8; ++j) acc[a][j] = 0.f;

    for (int k0 = 0; k0 < DD; k0 += 32) {
        v16bf bfrag[2];
#pragma unroll
        for (int ct = 0; ct < 2; ++ct) {
            const __bf16* bp = WfT + (size_t)(h_base + ct * 16 + lmod) * DD + k0 + kbB;
            bfrag[ct] = load_frag16(bp, bp + 16);
        }
#pragma unroll
        for (int rt = 0; rt < 4; ++rt) {
            const __bf16* ap = imgbf + ((size_t)b * NN + n0 + rt * 16 + lmod) * DD + k0 + kbA;
            v16bf af = load_frag16(ap, ap + 16);
#pragma unroll
            for (int ct = 0; ct < 2; ++ct)
                acc[rt * 2 + ct] = __builtin_amdgcn_wmma_f32_16x16x32_bf16(
                    false, af, false, bfrag[ct], (short)0, acc[rt * 2 + ct], false, false);
        }
    }

    const float2* posp = (const float2*)positions;
#pragma unroll
    for (int rt = 0; rt < 4; ++rt) {
        float px[8], py[8];
#pragma unroll
        for (int j = 0; j < 8; ++j) {
            float2 p = posp[(size_t)b * NN + n0 + rt * 16 + lhalf * 8 + j];
            px[j] = p.x; py[j] = p.y;
        }
#pragma unroll
        for (int ct = 0; ct < 2; ++ct) {
            int h = h_base + ct * 16 + lmod;
            float bias = b_feat[h];
            float w0 = W_pos[h];
            float w1 = W_pos[HH + h];
            v8bf hv;
#pragma unroll
            for (int j = 0; j < 8; ++j) {
                int m = rt * 16 + lhalf * 8 + j;
                float val = acc[rt * 2 + ct][j] + bias + px[j] * w0 + py[j] * w1;
                featbf[((size_t)b * NN + n0 + m) * HH + h] = (__bf16)val;
                hv[j] = (__bf16)fmaxf(val, 0.f);
            }
            *(v8bf*)(hT + ((size_t)b * HH + h) * NN + n0 + rt * 16 + lhalf * 8) = hv;
        }
    }
}

// ---------------------------------------------------------------------------
// K4: adjacency GEMM  agg = (A @ relu(feat)) / deg  per slide.
// Batch positions staged in LDS (32KB). The 64x32 A-tile (4 fragments) is
// synthesized ONCE per block per k-step by all 256 threads into a
// double-buffered bank-padded LDS tile (1 barrier/k-step); every wave then
// ds_load_b128's its fragments. Degrees accumulate in per-thread registers.
// ---------------------------------------------------------------------------
__global__ void gemm_adj_kernel(const float* __restrict__ positions,
                                const __bf16* __restrict__ hT,
                                __bf16* __restrict__ aggbf) {
    __shared__ float4 pos_s4[NN / 2];                 // 32KB: batch positions
    __shared__ __bf16 atile[2][4][32 * ALANE];        // double-buffered A tiles
    __shared__ float  deg_s[64];

    int cb = blockIdx.x & 1;
    int rb = blockIdx.x >> 1;
    int b  = rb >> 6;
    int n0 = (rb & 63) << 6;
    int tid = threadIdx.x;
    int wave = tid >> 5, lane = tid & 31;
    int lmod = lane & 15, lhalf = lane >> 4;
    int kbB = lhalf * 8;
    int h_base = (cb << 8) + wave * 32;

    // stage positions for this batch
    const float4* pg = (const float4*)(positions + (size_t)b * NN * 2);
    for (int i = tid; i < NN / 2; i += 256) pos_s4[i] = pg[i];
    if (tid < 64) deg_s[tid] = 0.f;
    __syncthreads();

    // synth-thread ownership: 8 contiguous k-slots of one fragment lane
    int sf    = tid >> 6;                 // fragment (row tile) 0..3
    int slane = (tid >> 1) & 31;          // fragment lane 0..31
    int sh    = tid & 1;                  // which 8-element half
    int srow  = sf * 16 + (slane & 15);   // row within 64-row stripe
    int skb   = ((slane >> 4) << 3) + (sh << 4);   // k_local base: 0/8/16/24
    const float2* pos_s = (const float2*)pos_s4;
    float2 pn = pos_s[n0 + srow];
    int rowG = n0 + srow;                 // within-batch row index
    float degsum = 0.f;

    v8f acc[8];
#pragma unroll
    for (int a = 0; a < 8; ++a)
#pragma unroll
        for (int j = 0; j < 8; ++j) acc[a][j] = 0.f;

    for (int m0 = 0; m0 < NN; m0 += 32) {
        int p = (m0 >> 5) & 1;
        // cooperative A synthesis: 8 elements per thread
        {
            v8bf av;
            int mbase = m0 + skb;
#pragma unroll
            for (int q = 0; q < 4; ++q) {
                float4 pp = pos_s4[(mbase >> 1) + q];
                int m1 = mbase + 2 * q, m2 = m1 + 1;
                float dx = pn.x - pp.x, dy = pn.y - pp.y;
                float ex = pn.x - pp.z, ey = pn.y - pp.w;
                float a1 = ((dx * dx + dy * dy) < R2C && m1 != rowG) ? 1.f : 0.f;
                float a2 = ((ex * ex + ey * ey) < R2C && m2 != rowG) ? 1.f : 0.f;
                av[2 * q]     = (__bf16)a1;
                av[2 * q + 1] = (__bf16)a2;
                degsum += a1 + a2;
            }
            *(v8bf*)&atile[p][sf][slane * ALANE + sh * 8] = av;
        }
        __syncthreads();
        // B fragments for this wave's 2 col tiles
        v16bf bfrag[2];
#pragma unroll
        for (int ct = 0; ct < 2; ++ct) {
            const __bf16* bp = hT + ((size_t)b * HH + h_base + ct * 16 + lmod) * NN + m0 + kbB;
            bfrag[ct] = load_frag16(bp, bp + 16);
        }
#pragma unroll
        for (int rt = 0; rt < 4; ++rt) {
            const __bf16* ap = &atile[p][rt][lane * ALANE];
            v16bf af = load_frag16(ap, ap + 8);
#pragma unroll
            for (int ct = 0; ct < 2; ++ct)
                acc[rt * 2 + ct] = __builtin_amdgcn_wmma_f32_16x16x32_bf16(
                    false, af, false, bfrag[ct], (short)0, acc[rt * 2 + ct], false, false);
        }
    }
    atomicAdd(&deg_s[srow], degsum);
    __syncthreads();

#pragma unroll
    for (int rt = 0; rt < 4; ++rt)
#pragma unroll
        for (int ct = 0; ct < 2; ++ct) {
            int h = h_base + ct * 16 + lmod;
#pragma unroll
            for (int j = 0; j < 8; ++j) {
                int m = rt * 16 + lhalf * 8 + j;
                float invd = 1.0f / fmaxf(deg_s[m], 1.0f);
                aggbf[((size_t)b * NN + n0 + m) * HH + h] = (__bf16)(acc[rt * 2 + ct][j] * invd);
            }
        }
}

// ---------------------------------------------------------------------------
// K5: refined = relu(agg@W_gnn + feat@W_self); column sums -> latent_acc.
// Same 64x256 blocking; two WMMA chains share one accumulator per tile.
// ---------------------------------------------------------------------------
__global__ void gemm_out_kernel(const __bf16* __restrict__ aggbf,
                                const __bf16* __restrict__ featbf,
                                const __bf16* __restrict__ WgT,
                                const __bf16* __restrict__ WsT,
                                float* __restrict__ latent_acc) {
    int cb = blockIdx.x & 1;
    int rb = blockIdx.x >> 1;
    int b  = rb >> 6;
    int n0 = (rb & 63) << 6;
    int tid = threadIdx.x;
    int wave = tid >> 5, lane = tid & 31;
    int lmod = lane & 15, lhalf = lane >> 4;
    int kbA = lhalf * 8;
    int kbB = lhalf * 8;
    int h_base = (cb << 8) + wave * 32;

    v8f acc[8];
#pragma unroll
    for (int a = 0; a < 8; ++a)
#pragma unroll
        for (int j = 0; j < 8; ++j) acc[a][j] = 0.f;

    for (int k0 = 0; k0 < HH; k0 += 32) {
        v16bf b1[2], b2[2];
#pragma unroll
        for (int ct = 0; ct < 2; ++ct) {
            const __bf16* bp1 = WgT + (size_t)(h_base + ct * 16 + lmod) * HH + k0 + kbB;
            const __bf16* bp2 = WsT + (size_t)(h_base + ct * 16 + lmod) * HH + k0 + kbB;
            b1[ct] = load_frag16(bp1, bp1 + 16);
            b2[ct] = load_frag16(bp2, bp2 + 16);
        }
#pragma unroll
        for (int rt = 0; rt < 4; ++rt) {
            const __bf16* ap1 = aggbf  + ((size_t)b * NN + n0 + rt * 16 + lmod) * HH + k0 + kbA;
            const __bf16* ap2 = featbf + ((size_t)b * NN + n0 + rt * 16 + lmod) * HH + k0 + kbA;
            v16bf a1 = load_frag16(ap1, ap1 + 16);
            v16bf a2 = load_frag16(ap2, ap2 + 16);
#pragma unroll
            for (int ct = 0; ct < 2; ++ct) {
                acc[rt * 2 + ct] = __builtin_amdgcn_wmma_f32_16x16x32_bf16(
                    false, a1, false, b1[ct], (short)0, acc[rt * 2 + ct], false, false);
                acc[rt * 2 + ct] = __builtin_amdgcn_wmma_f32_16x16x32_bf16(
                    false, a2, false, b2[ct], (short)0, acc[rt * 2 + ct], false, false);
            }
        }
    }
#pragma unroll
    for (int ct = 0; ct < 2; ++ct) {
        int h = h_base + ct * 16 + lmod;
        float s = 0.f;
#pragma unroll
        for (int rt = 0; rt < 4; ++rt)
#pragma unroll
            for (int j = 0; j < 8; ++j) s += fmaxf(acc[rt * 2 + ct][j], 0.f);
        atomicAdd(&latent_acc[b * HH + h], s);
    }
}

// ---------------------------------------------------------------------------
// K6: finalize: divide by N, concatenate [latent | concept]
// ---------------------------------------------------------------------------
__global__ void finalize_kernel(const float* __restrict__ latent_acc,
                                const float* __restrict__ concept_acc,
                                float* __restrict__ out) {
    int idx = blockIdx.x * blockDim.x + threadIdx.x;
    int total = BB * (HH + KK);
    if (idx >= total) return;
    int b = idx / (HH + KK), j = idx % (HH + KK);
    float v = (j < HH) ? latent_acc[b * HH + j] : concept_acc[b * KK + (j - HH)];
    out[idx] = v * (1.0f / (float)NN);
}

// ---------------------------------------------------------------------------
extern "C" void kernel_launch(void* const* d_in, const int* in_sizes, int n_in,
                              void* d_out, int out_size, void* d_ws, size_t ws_size,
                              hipStream_t stream) {
    const float* images    = (const float*)d_in[0];
    const float* positions = (const float*)d_in[1];
    const float* W_feat    = (const float*)d_in[2];
    const float* b_feat    = (const float*)d_in[3];
    const float* W_pos     = (const float*)d_in[4];
    const float* W_gnn     = (const float*)d_in[5];
    const float* W_self    = (const float*)d_in[6];
    const float* anchors   = (const float*)d_in[7];
    float* out = (float*)d_out;

    char* ws = (char*)d_ws;
    size_t off = 0;
    auto alloc = [&](size_t bytes) { char* p = ws + off; off += (bytes + 255) & ~(size_t)255; return p; };

    __bf16* img_bf = (__bf16*)alloc((size_t)BB * NN * DD * 2);
    __bf16* featbf = (__bf16*)alloc((size_t)BB * NN * HH * 2);
    __bf16* hT     = (__bf16*)alloc((size_t)BB * NN * HH * 2);
    __bf16* aggbf  = (__bf16*)alloc((size_t)BB * NN * HH * 2);
    __bf16* WfT    = (__bf16*)alloc((size_t)HH * DD * 2);
    __bf16* WgT    = (__bf16*)alloc((size_t)HH * HH * 2);
    __bf16* WsT    = (__bf16*)alloc((size_t)HH * HH * 2);
    float* latent_acc  = (float*)alloc((size_t)BB * HH * 4);
    float* concept_acc = (float*)alloc((size_t)BB * KK * 4);
    float* anchor_invn = (float*)alloc((size_t)KK * 4);

    init_kernel<<<1, 256, 0, stream>>>(anchors, anchor_invn, latent_acc, concept_acc);
    concept_cvt_kernel<<<BB * NN, 256, 0, stream>>>(images, anchors, anchor_invn,
                                                    img_bf, concept_acc);
    cvt_weights_kernel<<<(HH * DD + 255) / 256, 256, 0, stream>>>(W_feat, W_gnn, W_self,
                                                                  WfT, WgT, WsT);
    // 64-row x 256-col blocks, 2 col-blocks covering H=512
    gemm_feat_kernel<<<(BB * NN / 64) * 2, 256, 0, stream>>>(img_bf, WfT, b_feat, W_pos,
                                                             positions, featbf, hT);
    gemm_adj_kernel<<<(BB * NN / 64) * 2, 256, 0, stream>>>(positions, hT, aggbf);
    gemm_out_kernel<<<(BB * NN / 64) * 2, 256, 0, stream>>>(aggbf, featbf, WgT, WsT,
                                                            latent_acc);
    finalize_kernel<<<(BB * (HH + KK) + 255) / 256, 256, 0, stream>>>(latent_acc,
                                                                      concept_acc, out);
}
